// MLAAttention_21354577395889
// MI455X (gfx1250) — matrix operand order repaired and checked by reading, not compiled, not run on hardware
//
#include <hip/hip_runtime.h>
#include <math.h>

// ---- problem constants --------------------------------------------------
#define H_    32
#define R_    512
#define NOPE_ 128
#define ROPE_ 64
#define DQK_  576               // R_ + ROPE_
#define V_    128
#define HID_  4096
#define B_    64
#define S_    4096
#define TOPK_ 2048
#define QCOLS_ (H_ * (NOPE_ + ROPE_))   // 6144
#define SCALE_F 0.07216878364870323f    // (NOPE+ROPE)^-0.5 = 1/sqrt(192)

typedef __bf16 bf16;
typedef __attribute__((ext_vector_type(16))) __bf16 v16bf;
typedef __attribute__((ext_vector_type(8)))  __bf16 v8bf;
typedef __attribute__((ext_vector_type(4)))  __bf16 v4bf;
typedef __attribute__((ext_vector_type(2)))  __bf16 v2bf;
typedef __attribute__((ext_vector_type(8)))  float  v8f;

#define WMMA_BF16(a, b, c) \
    __builtin_amdgcn_wmma_f32_16x16x32_bf16(false, (a), false, (b), (short)0, (c), false, false)

// ---- async global->LDS (gfx1250 GLOBAL_LOAD_ASYNC_TO_LDS_B128, GV mode) --
__device__ __forceinline__ void async_ld_b128(void* lds_dst, const void* gsrc) {
    asm volatile("global_load_async_to_lds_b128 %0, %1, off"
                 :: "v"((unsigned)(uintptr_t)lds_dst),        // low 32b of generic = LDS offset
                    "v"((unsigned long long)(uintptr_t)gsrc)
                 : "memory");
}
__device__ __forceinline__ void wait_async0() {
    asm volatile("s_wait_asynccnt 0x0" ::: "memory");
}

// ---- 16x16x32 bf16 fragment loaders (wave32, contiguous k = 16h + s order) --
// A 16x32 bf16: lane m = lane&15; contiguous K 0..15 (+16 hi lanes)
__device__ __forceinline__ v16bf load_frag_a(const bf16* __restrict__ p, int ld) {
    const int lane = threadIdx.x & 31;
    const bf16* row = p + (lane & 15) * ld + ((lane & 16) ? 16 : 0);
    v8bf lo = *(const v8bf*)(row);        // K 0..7   (+koff)
    v8bf hi = *(const v8bf*)(row + 8);    // K 8..15  (+koff)
    return __builtin_shufflevector(lo, hi, 0, 1, 2, 3, 4, 5, 6, 7,
                                           8, 9, 10, 11, 12, 13, 14, 15);
}

// B 32x16 from row-major T[n][k] bf16 (logical B = T^T): one 32B run per lane
__device__ __forceinline__ v16bf load_frag_bt(const bf16* __restrict__ p, int ld) {
    const int lane = threadIdx.x & 31;
    const bf16* row = p + (lane & 15) * ld + ((lane & 16) ? 16 : 0);
    v8bf lo = *(const v8bf*)(row);
    v8bf hi = *(const v8bf*)(row + 8);
    return __builtin_shufflevector(lo, hi, 0, 1, 2, 3, 4, 5, 6, 7,
                                           8, 9, 10, 11, 12, 13, 14, 15);
}

// B 32x16 from row-major T[n][k] f32 source, convert to bf16 (contiguous f32)
__device__ __forceinline__ v16bf load_frag_bt_f32(const float* __restrict__ p, int ld) {
    const int lane = threadIdx.x & 31;
    const float* row = p + (lane & 15) * ld + ((lane & 16) ? 16 : 0);
    v16bf f;
#pragma unroll
    for (int j = 0; j < 16; ++j) f[j] = (bf16)row[j];
    return f;
}

__device__ __forceinline__ v4bf cvt4(float4 v) {
    return (v4bf){(bf16)v.x, (bf16)v.y, (bf16)v.z, (bf16)v.w};
}

// ---- Kernel: C[64][N] = A[64][K] @ B[K][N], all row-major f32, bf16 WMMA --
__global__ __launch_bounds__(256) void gemm_rm_kernel(
    const float* __restrict__ A, const float* __restrict__ Bm,
    float* __restrict__ C, int N, int K) {
    __shared__ bf16 As[64][40];   // A tile 64x32 (+pad)
    __shared__ bf16 Bs[64][40];   // B tile stored TRANSPOSED: Bs[n][k]
    const int n0   = blockIdx.x * 64;
    const int tid  = threadIdx.x;
    const int lane = tid & 31;
    const int wave = tid >> 5;
    const int mt   = wave >> 1;
    const int nt0  = (wave & 1) * 2;
    v8f acc0 = {}; v8f acc1 = {};

    for (int k0 = 0; k0 < K; k0 += 32) {
        __syncthreads();
#pragma unroll
        for (int r = 0; r < 2; ++r) {                 // A tile: 512 float4
            int idx4 = tid * 2 + r;
            int row = idx4 >> 3, c4 = idx4 & 7;
            float4 v = *(const float4*)(A + (size_t)row * K + k0 + 4 * c4);
            *(v4bf*)(&As[row][4 * c4]) = cvt4(v);
        }
#pragma unroll
        for (int r = 0; r < 2; ++r) {                 // B tile: 512 float4, transpose
            int idx4 = tid * 2 + r;
            int row = idx4 >> 4, c4 = idx4 & 15;      // row=k (32), c4 over 64 cols
            float4 v = *(const float4*)(Bm + (size_t)(k0 + row) * N + n0 + 4 * c4);
            Bs[4 * c4 + 0][row] = (bf16)v.x;
            Bs[4 * c4 + 1][row] = (bf16)v.y;
            Bs[4 * c4 + 2][row] = (bf16)v.z;
            Bs[4 * c4 + 3][row] = (bf16)v.w;
        }
        if (k0 + 32 < K)                               // prefetch next B tile
            __builtin_prefetch(Bm + (size_t)(k0 + 32) * N + n0 + lane * 8, 0, 0);
        __syncthreads();
        v16bf a  = load_frag_a(&As[mt * 16][0], 40);
        v16bf b0 = load_frag_bt(&Bs[nt0 * 16][0], 40);
        v16bf b1 = load_frag_bt(&Bs[(nt0 + 1) * 16][0], 40);
        acc0 = WMMA_BF16(a, b0, acc0);
        acc1 = WMMA_BF16(a, b1, acc1);
    }
    const int n  = lane & 15;
    const int mb = (lane & 16) ? 8 : 0;
#pragma unroll
    for (int e = 0; e < 8; ++e) {
        C[(size_t)(mt * 16 + mb + e) * N + n0 + nt0 * 16 + n]       = acc0[e];
        C[(size_t)(mt * 16 + mb + e) * N + n0 + (nt0 + 1) * 16 + n] = acc1[e];
    }
}

// ---- Kernel: per-head C[64][.] = A[64][K] @ T[n][K]^T * scale ------------
__global__ __launch_bounds__(256) void gemm_abt_kernel(
    const float* __restrict__ A, int lda, int aHeadStride,
    const float* __restrict__ T, int ldt, int nPerHead,
    float* __restrict__ C, int ldc, int cHeadStride,
    int K, const float* __restrict__ scale_p, float extraScale) {
    __shared__ bf16 As[64][40];
    __shared__ bf16 Ts[64][40];
    const int h    = blockIdx.y;
    const int n0   = blockIdx.x * 64;
    const int tid  = threadIdx.x;
    const int lane = tid & 31;
    const int wave = tid >> 5;
    const int mt   = wave >> 1;
    const int nt0  = (wave & 1) * 2;
    const float* Ah = A + (size_t)h * aHeadStride;
    const float* Th = T + (size_t)h * nPerHead * K;
    float* Ch = C + (size_t)h * cHeadStride;
    v8f acc0 = {}; v8f acc1 = {};

    for (int k0 = 0; k0 < K; k0 += 32) {
        __syncthreads();
#pragma unroll
        for (int r = 0; r < 2; ++r) {
            int idx4 = tid * 2 + r;
            int row = idx4 >> 3, c4 = idx4 & 7;
            float4 va = *(const float4*)(Ah + (size_t)row * lda + k0 + 4 * c4);
            float4 vt = *(const float4*)(Th + (size_t)(n0 + row) * ldt + k0 + 4 * c4);
            *(v4bf*)(&As[row][4 * c4]) = cvt4(va);
            *(v4bf*)(&Ts[row][4 * c4]) = cvt4(vt);
        }
        if (k0 + 32 < K)
            __builtin_prefetch(Th + (size_t)n0 * ldt + k0 + 32 + lane * 8, 0, 0);
        __syncthreads();
        v16bf a  = load_frag_a(&As[mt * 16][0], 40);
        v16bf b0 = load_frag_bt(&Ts[nt0 * 16][0], 40);
        v16bf b1 = load_frag_bt(&Ts[(nt0 + 1) * 16][0], 40);
        acc0 = WMMA_BF16(a, b0, acc0);
        acc1 = WMMA_BF16(a, b1, acc1);
    }
    const float s  = scale_p[0] * extraScale;
    const int n  = lane & 15;
    const int mb = (lane & 16) ? 8 : 0;
#pragma unroll
    for (int e = 0; e < 8; ++e) {
        Ch[(size_t)(mt * 16 + mb + e) * ldc + n0 + nt0 * 16 + n]       = acc0[e] * s;
        Ch[(size_t)(mt * 16 + mb + e) * ldc + n0 + (nt0 + 1) * 16 + n] = acc1[e] * s;
    }
}

// ---- Kernel: RoPE on q_pe -> qfull[b][h][512:576] (pre-scaled) ----------
__global__ __launch_bounds__(1024) void rope_kernel(
    const float* __restrict__ q,          // [B][6144]
    const int* __restrict__ positions,    // [B]
    float* __restrict__ qfull) {          // [B][H][576]
    const int b = blockIdx.x;
    const int h = threadIdx.x >> 5;
    const int i = threadIdx.x & 31;       // half = 32
    const float pos  = (float)positions[b];
    const float invf = exp2f(-((float)(2 * i) / 64.0f) * 13.287712379549449f); // log2(1e4)
    const float fr = pos * invf;
    const float c = cosf(fr), s = sinf(fr);
    const float* qp = q + (size_t)b * QCOLS_ + h * (NOPE_ + ROPE_) + NOPE_;
    const float q1 = qp[i], q2 = qp[32 + i];
    float* dst = qfull + ((size_t)b * H_ + h) * DQK_ + R_;
    dst[i]      = (q1 * c - q2 * s) * SCALE_F;
    dst[32 + i] = (q2 * c + q1 * s) * SCALE_F;
}

// ---- Kernel: fused sparse flash-attention (1 block per batch) -----------
// LDS layout (bytes):
//   q_lds   bf16 32x592          @ 0        37,888
//   kv_lds  f32  64x592          @ 37,888   151,552   (async gather target)
//   kvt_lds bf16 512x72          @ 189,440  73,728    (c_kv transposed, bf16)
//   sc_lds  f32  32x72           @ 263,168  9,216
//   p_lds   bf16 32x72           @ 272,384  4,608
//   m/l/a   f32  3x32            @ 276,992  384       total = 277,376
__global__ __launch_bounds__(256) void mla_attn_kernel(
    const float* __restrict__ qfull,   // [B][H][576], pre-scaled
    const float* __restrict__ kv,      // [B][S][576]
    const int*   __restrict__ topk,    // [B][TOPK]
    float* __restrict__ o_out) {       // [B][H][512]
    extern __shared__ char smem[];
    const int QSTR  = 592;   // bf16 q stride
    const int QSTRF = 592;   // f32 kv stride
    const int TSTR  = 72;    // kvt token stride
    bf16*  q_lds   = (bf16*)smem;
    float* kv_lds  = (float*)(smem + 37888);
    bf16*  kvt_lds = (bf16*)(smem + 189440);
    float* sc_lds  = (float*)(smem + 263168);
    bf16*  p_lds   = (bf16*)(smem + 272384);
    float* mrow    = (float*)(smem + 276992);
    float* lrow    = mrow + 32;
    float* arow    = lrow + 32;

    const int b    = blockIdx.x;
    const int tid  = threadIdx.x;
    const int lane = tid & 31;
    const int wave = tid >> 5;
    const int smt = wave & 1, snt = wave >> 1;   // score tile (heads, tokens)
    const int omt = wave & 1, ong = wave >> 1;   // o tile (heads, feature group)

    // stage q (f32 -> bf16 LDS), vectorized
    {
        const float* qb = qfull + (size_t)b * H_ * DQK_;
        for (int hh = 0; hh < H_; ++hh)
            for (int c4 = tid; c4 < DQK_ / 4; c4 += 256) {
                float4 v = *(const float4*)(qb + hh * DQK_ + 4 * c4);
                *(v4bf*)(q_lds + hh * QSTR + 4 * c4) = cvt4(v);
            }
    }
    if (tid < 32) { mrow[tid] = -1e30f; lrow[tid] = 0.0f; }
    v8f oacc[8];
#pragma unroll
    for (int i = 0; i < 8; ++i) oacc[i] = (v8f){};

    for (int kt = 0; kt < TOPK_ / 64; ++kt) {
        __syncthreads();
        // ---- async gather: 64 selected kv rows (raw f32) -> LDS
        {
            const int row   = tid >> 2;
            const int cpart = (tid & 3) * 144;
            const int idx   = topk[(size_t)b * TOPK_ + kt * 64 + row];
            const float* src = kv + ((size_t)b * S_ + idx) * DQK_ + cpart;
            float* dst = kv_lds + row * QSTRF + cpart;
#pragma unroll
            for (int j = 0; j < 144; j += 4)
                async_ld_b128(dst + j, src + j);
            wait_async0();
        }
        __syncthreads();
        // ---- transpose/convert c_kv: kv_lds[token][feat<512] -> kvt[feat][token]
        {
            const int token = tid >> 2;
            const int fpart = (tid & 3) * 128;
            const float* srcr = kv_lds + token * QSTRF + fpart;
#pragma unroll
            for (int j = 0; j < 128; j += 4) {
                float4 v = *(const float4*)(srcr + j);
                kvt_lds[(fpart + j + 0) * TSTR + token] = (bf16)v.x;
                kvt_lds[(fpart + j + 1) * TSTR + token] = (bf16)v.y;
                kvt_lds[(fpart + j + 2) * TSTR + token] = (bf16)v.z;
                kvt_lds[(fpart + j + 3) * TSTR + token] = (bf16)v.w;
            }
        }
        // ---- scores: (32 heads x 64 tokens) over K = 576 (reads kv_lds f32)
        {
            v8f sacc = {};
            const bf16*  ap = q_lds + (smt * 16) * QSTR;
            const float* bp = kv_lds + (snt * 16) * QSTRF;
#pragma unroll
            for (int kk = 0; kk < DQK_; kk += 32) {
                v16bf af  = load_frag_a(ap + kk, QSTR);
                v16bf bf_ = load_frag_bt_f32(bp + kk, QSTRF);
                sacc = WMMA_BF16(af, bf_, sacc);
            }
            const int n  = lane & 15;
            const int mb = (lane & 16) ? 8 : 0;
#pragma unroll
            for (int e = 0; e < 8; ++e)
                sc_lds[(smt * 16 + mb + e) * 72 + snt * 16 + n] = sacc[e];
        }
        __syncthreads();
        // ---- online softmax bookkeeping (one lane per head)
        if (tid < 32) {
            const int hh = tid;
            float mo = mrow[hh];
            float tmax = -1e30f;
            for (int j = 0; j < 64; ++j) tmax = fmaxf(tmax, sc_lds[hh * 72 + j]);
            float mn = fmaxf(mo, tmax);
            float al = __expf(mo - mn);
            float ssum = 0.0f;
            for (int j = 0; j < 64; ++j) {
                float p = __expf(sc_lds[hh * 72 + j] - mn);
                ssum += p;
                p_lds[hh * 72 + j] = (bf16)p;
            }
            mrow[hh] = mn;
            lrow[hh] = lrow[hh] * al + ssum;
            arow[hh] = al;
        }
        __syncthreads();
        // ---- rescale o and accumulate p @ c_kv (B frags from bf16 kvt, vectorized)
        {
            const int mb = (lane & 16) ? 8 : 0;
            float alv[8];
#pragma unroll
            for (int e = 0; e < 8; ++e) alv[e] = arow[omt * 16 + mb + e];
#pragma unroll
            for (int i = 0; i < 8; ++i)
#pragma unroll
                for (int e = 0; e < 8; ++e) oacc[i][e] *= alv[e];
#pragma unroll
            for (int kk2 = 0; kk2 < 2; ++kk2) {
                v16bf af = load_frag_a(p_lds + (omt * 16) * 72 + kk2 * 32, 72);
#pragma unroll
                for (int i = 0; i < 8; ++i) {
                    v16bf bf_ = load_frag_bt(
                        kvt_lds + (ong * 128 + i * 16) * TSTR + kk2 * 32, TSTR);
                    oacc[i] = WMMA_BF16(af, bf_, oacc[i]);
                }
            }
        }
    }
    __syncthreads();
    // ---- normalize by l and write o
    {
        const int mb = (lane & 16) ? 8 : 0;
        const int n  = lane & 15;
        float inv[8];
#pragma unroll
        for (int e = 0; e < 8; ++e) inv[e] = 1.0f / lrow[omt * 16 + mb + e];
#pragma unroll
        for (int i = 0; i < 8; ++i)
#pragma unroll
            for (int e = 0; e < 8; ++e)
                o_out[((size_t)b * H_ + omt * 16 + mb + e) * R_ + ong * 128 + i * 16 + n] =
                    oacc[i][e] * inv[e];
    }
}

// ---- host side ----------------------------------------------------------
extern "C" void kernel_launch(void* const* d_in, const int* in_sizes, int n_in,
                              void* d_out, int out_size, void* d_ws, size_t ws_size,
                              hipStream_t stream) {
    const float* x         = (const float*)d_in[0];
    const float* Wq        = (const float*)d_in[1];
    const float* W_K       = (const float*)d_in[2];   // [H][R][NOPE]
    const float* W_K_scale = (const float*)d_in[3];
    const float* W_V       = (const float*)d_in[4];   // [H][V][R]
    const float* W_V_scale = (const float*)d_in[5];
    const float* Wo        = (const float*)d_in[6];
    const float* kv        = (const float*)d_in[7];
    const int*   topk      = (const int*)d_in[8];
    const int*   pos       = (const int*)d_in[9];
    float*       out       = (float*)d_out;

    float* q_ws   = (float*)d_ws;                   // [B][6144]
    float* qfull  = q_ws + (size_t)B_ * QCOLS_;     // [B][H][576]
    float* o_ws   = qfull + (size_t)B_ * H_ * DQK_; // [B][H][512]
    float* tmp_ws = o_ws + (size_t)B_ * H_ * R_;    // [B][4096]

    // 1) q = x @ Wq
    gemm_rm_kernel<<<dim3(QCOLS_ / 64), 256, 0, stream>>>(x, Wq, q_ws, QCOLS_, HID_);
    // 2) RoPE -> qfull[..., 512:576] (pre-scaled)
    rope_kernel<<<dim3(B_), dim3(1024), 0, stream>>>(q_ws, pos, qfull);
    // 3) ql = q_nope @ W_K^T * (W_K_scale * SCALE) -> qfull[..., 0:512]
    gemm_abt_kernel<<<dim3(R_ / 64, H_), 256, 0, stream>>>(
        q_ws, QCOLS_, NOPE_ + ROPE_, W_K, NOPE_, R_,
        qfull, H_ * DQK_, DQK_, NOPE_, W_K_scale, SCALE_F);
    // 4) fused sparse attention (async-gather + in-LDS transpose + WMMA flash loop)
    constexpr size_t SMEM = 277376;
    mla_attn_kernel<<<dim3(B_), 256, SMEM, stream>>>(qfull, kv, topk, o_ws);
    // 5) tmp = o @ W_V^T * W_V_scale
    gemm_abt_kernel<<<dim3(V_ / 64, H_), 256, 0, stream>>>(
        o_ws, H_ * R_, R_, W_V, R_, V_,
        tmp_ws, HID_, V_, R_, W_V_scale, 1.0f);
    // 6) out = tmp @ Wo
    gemm_rm_kernel<<<dim3(HID_ / 64), 256, 0, stream>>>(tmp_ws, Wo, out, HID_, HID_);
}
